// GraphSAGENodePredictor_68453188764197
// MI455X (gfx1250) — matrix operand order repaired and checked by reading, not compiled, not run on hardware
//
#include <hip/hip_runtime.h>
#include <hip/hip_bf16.h>
#include <math.h>

#define N_NODES 100000
#define N_EDGES 1200000
#define DIM     64
#define HID2    32
#define BN_EPS  1e-5f

typedef __attribute__((ext_vector_type(16))) __bf16 v16bf;
typedef __attribute__((ext_vector_type(8)))  float  v8f;

// native f32 -> bf16 conversion (backend selects hw cvt on gfx1250)
__device__ __forceinline__ __bf16 f2bf(float f) { return (__bf16)f; }

__global__ void zero_f32(float* __restrict__ p, int n) {
    int i = blockIdx.x * blockDim.x + threadIdx.x;
    if (i < n) p[i] = 0.0f;
}

__global__ void degree_kernel(const int* __restrict__ dst, float* __restrict__ cnt) {
    int e = blockIdx.x * blockDim.x + threadIdx.x;
    if (e < N_EDGES) atomicAdd(&cnt[dst[e]], 1.0f);
}

// one thread handles 4 consecutive features of one edge: float4 gather + 4 f32 atomics
// (77MB working set is L2-resident on 192MB L2 -> atomics resolve at L2 bandwidth)
__global__ void scatter_kernel(const float* __restrict__ feat,
                               const int* __restrict__ src,
                               const int* __restrict__ dst,
                               float* __restrict__ agg) {
    long long gid = (long long)blockIdx.x * blockDim.x + threadIdx.x;
    if (gid >= (long long)N_EDGES * 16) return;
    int e = (int)(gid >> 4);
    if ((gid & 15) == 0 && e + 4096 < N_EDGES) {
        __builtin_prefetch(src + e + 4096, 0, 0);   // global_prefetch_b8
        __builtin_prefetch(dst + e + 4096, 0, 0);
    }
    int q = ((int)gid & 15) * 4;
    int s = src[e], d = dst[e];
    float4 vsrc = *(const float4*)(feat + (long long)s * DIM + q);
    float* dp = agg + (long long)d * DIM + q;
    atomicAdd(dp + 0, vsrc.x);
    atomicAdd(dp + 1, vsrc.y);
    atomicAdd(dp + 2, vsrc.z);
    atomicAdd(dp + 3, vsrc.w);
}

// Load 16x32 bf16 A-fragment for this lane (contiguous k order, same as B-frag):
// lane = half*16 + m;  element e -> K = kstep*32 + 16*half + e
__device__ __forceinline__ v16bf load_a_frag(const float* row, int kstep, int h, float scale) {
    v16bf a;
    const float* p0 = row + kstep * 32 + h * 16;       // K = kstep*32 + 16h + (0..7)
    const float* p1 = p0 + 8;                          // K = kstep*32 + 16h + 8 + (0..7)
#pragma unroll
    for (int i = 0; i < 8; ++i) a[i] = f2bf(p0[i] * scale);
#pragma unroll
    for (int i = 0; i < 8; ++i) a[i + 8] = f2bf(p1[i] * scale);
    return a;
}

// Fused SAGE layer: out = relu( BN( (agg/cnt)@Wl.T + bl + x@Wr.T ) )
// 8 waves/block, each wave owns a 16-row x 64-col tile -> 16 v_wmma_f32_16x16x32_bf16.
// NOTE: out may alias aggs (layer 2 in-place): each tile's aggs rows are fully
// consumed into registers before any store (true data dependency), and no other
// wave touches those rows.
__global__ void __launch_bounds__(256)
sage_layer_kernel(const float* __restrict__ xin,
                  const float* aggs,            // may alias out
                  const float* __restrict__ cnt,
                  const float* __restrict__ Wl, const float* __restrict__ bl,
                  const float* __restrict__ Wr,
                  const float* __restrict__ g,  const float* __restrict__ beta,
                  const float* __restrict__ m,  const float* __restrict__ v,
                  float* out) {
    __shared__ __attribute__((aligned(32))) __bf16 WlB[DIM * DIM];
    __shared__ __attribute__((aligned(32))) __bf16 WrB[DIM * DIM];
    __shared__ float s_scale[DIM], s_shift[DIM];

    int tid = threadIdx.x;
    for (int i = tid; i < DIM * DIM; i += 256) {
        WlB[i] = f2bf(Wl[i]);                  // row-major [n][k], matches B-frag reads
        WrB[i] = f2bf(Wr[i]);
    }
    if (tid < DIM) {
        float s = g[tid] * rsqrtf(v[tid] + BN_EPS);
        s_scale[tid] = s;
        s_shift[tid] = beta[tid] + (bl[tid] - m[tid]) * s;  // fold bias + BN shift
    }
    __syncthreads();

    int wave = tid >> 5, lane = tid & 31;
    int row0 = (blockIdx.x * 8 + wave) * 16;
    if (row0 >= N_NODES) return;               // wave-uniform: EXEC stays all-ones

    int mloc = lane & 15, h = lane >> 4;
    int row  = row0 + mloc;
    float inv = 1.0f / fmaxf(cnt[row], 1.0f);

    const float* aggrow = aggs + (long long)row * DIM;
    const float* xrow   = xin  + (long long)row * DIM;
    v16bf a_agg0 = load_a_frag(aggrow, 0, h, inv);
    v16bf a_agg1 = load_a_frag(aggrow, 1, h, inv);
    v16bf a_x0   = load_a_frag(xrow,   0, h, 1.0f);
    v16bf a_x1   = load_a_frag(xrow,   1, h, 1.0f);

#pragma unroll
    for (int nt = 0; nt < 4; ++nt) {
        int n = nt * 16 + mloc;                // B-frag column for this lane
        // B-fragment: 16 contiguous bf16 at W[n][kstep*32 + 16*half] (32B-aligned LDS)
        const v16bf* bl0 = (const v16bf*)&WlB[n * DIM +  0 + h * 16];
        const v16bf* bl1 = (const v16bf*)&WlB[n * DIM + 32 + h * 16];
        const v16bf* br0 = (const v16bf*)&WrB[n * DIM +  0 + h * 16];
        const v16bf* br1 = (const v16bf*)&WrB[n * DIM + 32 + h * 16];

        v8f acc = {};
        acc = __builtin_amdgcn_wmma_f32_16x16x32_bf16(false, a_agg0, false, *bl0, (short)0, acc, false, false);
        acc = __builtin_amdgcn_wmma_f32_16x16x32_bf16(false, a_agg1, false, *bl1, (short)0, acc, false, false);
        acc = __builtin_amdgcn_wmma_f32_16x16x32_bf16(false, a_x0,   false, *br0, (short)0, acc, false, false);
        acc = __builtin_amdgcn_wmma_f32_16x16x32_bf16(false, a_x1,   false, *br1, (short)0, acc, false, false);

        // C/D layout: col j = nt*16 + lane%16, row = r + 8*(lane/16)
        float sc = s_scale[n], sh = s_shift[n];
#pragma unroll
        for (int r = 0; r < 8; ++r) {
            int orow = row0 + r + 8 * h;
            float val = acc[r] * sc + sh;
            out[(long long)orow * DIM + n] = fmaxf(val, 0.0f);
        }
    }
}

// MLP head via WMMA: hid = relu(h2@mW1.T + mb1)  [16 rows x 32 cols per wave,
// 2 n-tiles x 2 k-steps = 4 WMMAs], then the 32->1 dot is a per-half shuffle
// reduction in C-layout (row m = r + 8*half lives across the 16 lanes of one half).
__global__ void __launch_bounds__(256)
mlp_head_kernel(const float* __restrict__ h2,
                const float* __restrict__ mW1, const float* __restrict__ mb1,
                const float* __restrict__ mW2, const float* __restrict__ mb2,
                float* __restrict__ outp) {
    __shared__ __attribute__((aligned(32))) __bf16 W1B[HID2 * DIM];
    __shared__ float sb1[HID2], sW2[HID2], sb2;
    int tid = threadIdx.x;
    for (int i = tid; i < HID2 * DIM; i += 256) W1B[i] = f2bf(mW1[i]);
    if (tid < HID2) { sb1[tid] = mb1[tid]; sW2[tid] = mW2[tid]; }
    if (tid == 0) sb2 = mb2[0];
    __syncthreads();

    int wave = tid >> 5, lane = tid & 31;
    int row0 = (blockIdx.x * 8 + wave) * 16;
    if (row0 >= N_NODES) return;               // wave-uniform
    int mloc = lane & 15, h = lane >> 4;
    int row  = row0 + mloc;

    const float* hrow = h2 + (long long)row * DIM;
    v16bf a0 = load_a_frag(hrow, 0, h, 1.0f);
    v16bf a1 = load_a_frag(hrow, 1, h, 1.0f);

    float zacc[8];
#pragma unroll
    for (int r = 0; r < 8; ++r) zacc[r] = 0.0f;

#pragma unroll
    for (int nt = 0; nt < 2; ++nt) {
        int n = nt * 16 + mloc;
        const v16bf* b0 = (const v16bf*)&W1B[n * DIM +  0 + h * 16];
        const v16bf* b1 = (const v16bf*)&W1B[n * DIM + 32 + h * 16];
        v8f acc = {};
        acc = __builtin_amdgcn_wmma_f32_16x16x32_bf16(false, a0, false, *b0, (short)0, acc, false, false);
        acc = __builtin_amdgcn_wmma_f32_16x16x32_bf16(false, a1, false, *b1, (short)0, acc, false, false);
        float bj = sb1[n], wj = sW2[n];
#pragma unroll
        for (int r = 0; r < 8; ++r)
            zacc[r] += fmaxf(acc[r] + bj, 0.0f) * wj;   // relu + weight of final dot
    }

    // reduce the 32-wide hidden dot across the 16 lanes of this half
#pragma unroll
    for (int r = 0; r < 8; ++r) {
        float t = zacc[r];
        t += __shfl_xor(t, 1, 16);
        t += __shfl_xor(t, 2, 16);
        t += __shfl_xor(t, 4, 16);
        t += __shfl_xor(t, 8, 16);
        if (mloc == r) {
            int orow = row0 + r + 8 * h;
            outp[orow] = 1.0f / (1.0f + expf(-(t + sb2)));
        }
    }
}

extern "C" void kernel_launch(void* const* d_in, const int* in_sizes, int n_in,
                              void* d_out, int out_size, void* d_ws, size_t ws_size,
                              hipStream_t stream) {
    const float* x   = (const float*)d_in[0];
    const int*   ei  = (const int*)d_in[1];     // [2][E] int32
    const float* Wl1 = (const float*)d_in[2];
    const float* bl1 = (const float*)d_in[3];
    const float* Wr1 = (const float*)d_in[4];
    const float* g1  = (const float*)d_in[5];
    const float* be1 = (const float*)d_in[6];
    const float* m1  = (const float*)d_in[7];
    const float* v1  = (const float*)d_in[8];
    const float* Wl2 = (const float*)d_in[9];
    const float* bl2 = (const float*)d_in[10];
    const float* Wr2 = (const float*)d_in[11];
    const float* g2  = (const float*)d_in[12];
    const float* be2 = (const float*)d_in[13];
    const float* m2  = (const float*)d_in[14];
    const float* v2  = (const float*)d_in[15];
    const float* mW1 = (const float*)d_in[16];
    const float* mb1 = (const float*)d_in[17];
    const float* mW2 = (const float*)d_in[18];
    const float* mb2 = (const float*)d_in[19];
    const int* src = ei;
    const int* dst = ei + N_EDGES;

    char* ws = (char*)d_ws;
    float* agg = (float*)ws;                                      // N*64 f32
    float* h1  = (float*)(ws + (size_t)N_NODES * DIM * 4);        // N*64 f32
    float* cnt = (float*)(ws + (size_t)N_NODES * DIM * 8);        // N   f32

    dim3 blk(256);
    int zgrid  = (N_NODES * DIM + 255) / 256;
    int cgrid  = (N_NODES + 255) / 256;
    int egrid  = (N_EDGES + 255) / 256;
    long long st = (long long)N_EDGES * 16;
    int sgrid  = (int)((st + 255) / 256);
    int lgrid  = (N_NODES + 127) / 128;   // 8 waves/block x 16 rows/wave

    // degrees (shared by both layers) + layer-1 aggregation
    zero_f32<<<cgrid, blk, 0, stream>>>(cnt, N_NODES);
    zero_f32<<<zgrid, blk, 0, stream>>>(agg, N_NODES * DIM);
    degree_kernel<<<egrid, blk, 0, stream>>>(dst, cnt);
    scatter_kernel<<<sgrid, blk, 0, stream>>>(x, src, dst, agg);
    sage_layer_kernel<<<lgrid, blk, 0, stream>>>(x, agg, cnt,
                                                 Wl1, bl1, Wr1, g1, be1, m1, v1, h1);
    // layer-2 aggregation over h1; linear writes in-place over agg
    zero_f32<<<zgrid, blk, 0, stream>>>(agg, N_NODES * DIM);
    scatter_kernel<<<sgrid, blk, 0, stream>>>(h1, src, dst, agg);
    sage_layer_kernel<<<lgrid, blk, 0, stream>>>(h1, agg, cnt,
                                                 Wl2, bl2, Wr2, g2, be2, m2, v2, agg);
    // MLP head + sigmoid (WMMA)
    mlp_head_kernel<<<lgrid, blk, 0, stream>>>(agg, mW1, mb1, mW2, mb2, (float*)d_out);
}
